// CausalMultiheadAttentionWrapper_68831145886620
// MI455X (gfx1250) — matrix operand order repaired and checked by reading, not compiled, not run on hardware
//
#include <hip/hip_runtime.h>
#include <math.h>
#include <stdint.h>

// ---- problem constants (match reference) ----
#define Bb 4
#define Tt 2048
#define Dm 1024
#define Hh 8
#define HD 128   // head dim
#define KSTEP 32
#define LPAD 40  // LDS row pitch in halves (20 dwords -> conflict-free row stride)

typedef _Float16 h8   __attribute__((ext_vector_type(8)));
typedef _Float16 v16h __attribute__((ext_vector_type(16)));
typedef float    v8f  __attribute__((ext_vector_type(8)));
typedef float    f4   __attribute__((ext_vector_type(4)));

// ---------------------------------------------------------------------------
// WMMA fragment layout helpers (wave32, v_wmma_f32_16x16x32_f16)
//
// A (16x32 MxK): lane L -> row m0+(L&15); halves = K {koff..koff+7, koff+16..23},
//   koff = k0 + (L>=16 ? 8 : 0)  -> two contiguous 16B loads.
// B (32x16 KxN) stored as Wt[N][K] row-major (computing A*Wt^T):
//   lane L -> col n0+(L&15); halves = K {kb..kb+7, kb+16..kb+23}, kb = k0 + (L>=16 ? 8 : 0)
//   -> two 16B loads from row n (same k order as A).
// C/D f32: VGPR r, lane L -> (m0 + r + (L>=16?8:0), n0 + (L&15)).
// ---------------------------------------------------------------------------

__device__ __forceinline__ v16h cat16(h8 lo, h8 hi) {
  return __builtin_shufflevector(lo, hi, 0,1,2,3,4,5,6,7,8,9,10,11,12,13,14,15);
}

__device__ __forceinline__ v16h load_a_f16(const _Float16* __restrict__ A, int lda,
                                           int m0, int k0, int lane) {
  const _Float16* p = A + (size_t)(m0 + (lane & 15)) * (size_t)lda
                        + (size_t)(k0 + ((lane >> 4) << 3));
  return cat16(*(const h8*)p, *(const h8*)(p + 16));
}

__device__ __forceinline__ v16h load_b_f16(const _Float16* __restrict__ Wt, int ldb,
                                           int n0, int k0, int lane) {
  const _Float16* p = Wt + (size_t)(n0 + (lane & 15)) * (size_t)ldb
                         + (size_t)(k0 + ((lane >> 4) << 3));
  return cat16(*(const h8*)p, *(const h8*)(p + 16));
}

__device__ __forceinline__ v16h load_a_f32(const float* __restrict__ P, int ldp,
                                           int m0, int k0, int lane) {
  const float* p = P + (size_t)(m0 + (lane & 15)) * (size_t)ldp
                     + (size_t)(k0 + ((lane >> 4) << 3));
  f4 a0 = *(const f4*)p;
  f4 a1 = *(const f4*)(p + 4);
  f4 b0 = *(const f4*)(p + 16);
  f4 b1 = *(const f4*)(p + 20);
  v16h r;
#pragma unroll
  for (int i = 0; i < 4; ++i) {
    r[i]      = (_Float16)a0[i];
    r[4 + i]  = (_Float16)a1[i];
    r[8 + i]  = (_Float16)b0[i];
    r[12 + i] = (_Float16)b1[i];
  }
  return r;
}

__device__ __forceinline__ v8f wmma16(v16h a, v16h b, v8f c) {
  return __builtin_amdgcn_wmma_f32_16x16x32_f16(false, a, false, b,
                                                (short)0, c, false, false);
}

// ---------------------------------------------------------------------------
// CDNA5 async global->LDS copy (ASYNCcnt-tracked), 16B per lane.
// VGLOBAL async encoding: VDST = LDS byte-offset VGPR, VADDR = 64-bit address.
// ---------------------------------------------------------------------------
__device__ __forceinline__ uint32_t lds_off(const void* p) {
  // generic (flat) shared pointer: low 32 bits are the LDS offset (aperture in hi32)
  return (uint32_t)(uintptr_t)p;
}

__device__ __forceinline__ void async_cp16(uint32_t lds, const _Float16* g) {
  asm volatile("global_load_async_to_lds_b128 %0, %1, off"
               :: "v"(lds), "v"((unsigned long long)(uintptr_t)g)
               : "memory");
}

__device__ __forceinline__ void wait_async0() {
#if __has_builtin(__builtin_amdgcn_s_wait_asynccnt)
  __builtin_amdgcn_s_wait_asynccnt(0);
#else
  asm volatile("s_wait_asynccnt 0x0" ::: "memory");
#endif
}

// ---------------------------------------------------------------------------
// Kernel 0: fp32 -> fp16 convert (grid-stride)
// ---------------------------------------------------------------------------
__global__ void __launch_bounds__(256)
f32_to_f16(const float* __restrict__ s, _Float16* __restrict__ d, size_t n) {
  size_t i = (size_t)blockIdx.x * blockDim.x + threadIdx.x;
  size_t stride = (size_t)gridDim.x * blockDim.x;
  for (; i < n; i += stride) d[i] = (_Float16)s[i];
}

// ---------------------------------------------------------------------------
// Kernel 1/4: C[M x N] = A16[M x K] * Wt16[N x K]^T + bias
// Double-buffered LDS pipeline: async copies for K-tile kt+1 overlap the
// WMMAs on tile kt; per-wave wait on own ASYNCcnt + s_barrier makes the
// whole tile visible. WG tile 64x128, 8 waves (4 M x 2 N), wave tile 16x64.
//   mode 0: f16 -> [B,H,T,hd]     (Q, K projections)
//   mode 2: f16 -> [B,H,hd,T]     (V^T)
//   mode 3: f32 -> [M x N]        (output projection)
// ---------------------------------------------------------------------------
__global__ void __launch_bounds__(256)
gemm_proj(const _Float16* __restrict__ A, const _Float16* __restrict__ Wt,
          const float* __restrict__ bias, _Float16* __restrict__ out_h,
          float* __restrict__ out_f, int N, int K, int mode) {
  __shared__ __attribute__((aligned(16))) _Float16 As[2][64][LPAD];
  __shared__ __attribute__((aligned(16))) _Float16 Bs[2][128][LPAD];

  const int t    = threadIdx.x;
  const int lane = t & 31;
  const int w    = t >> 5;
  const int mt   = blockIdx.x * 64;
  const int nt   = blockIdx.y * 128;
  const int m0l  = (w & 3) * 16;
  const int n0l  = (w >> 2) * 64;

  // per-thread copy assignment: 1 A chunk + 2 B chunks of 16B per K-tile
  const int ar = t >> 2;            // 0..63
  const int cc = (t & 3) * 8;       // half offset within 32-wide K tile
  const int br1 = ar + 64;          // B rows 64..127
  const _Float16* gA  = A  + (size_t)(mt + ar)  * (size_t)K + cc;
  const _Float16* gB0 = Wt + (size_t)(nt + ar)  * (size_t)K + cc;
  const _Float16* gB1 = Wt + (size_t)(nt + br1) * (size_t)K + cc;
  const uint32_t lA[2]  = { lds_off(&As[0][ar][cc]),  lds_off(&As[1][ar][cc])  };
  const uint32_t lB0[2] = { lds_off(&Bs[0][ar][cc]),  lds_off(&Bs[1][ar][cc])  };
  const uint32_t lB1[2] = { lds_off(&Bs[0][br1][cc]), lds_off(&Bs[1][br1][cc]) };

  auto issue = [&](int buf, int k0) {
    async_cp16(lA[buf],  gA  + k0);
    async_cp16(lB0[buf], gB0 + k0);
    async_cp16(lB1[buf], gB1 + k0);
  };

  v8f acc[4] = {};
  const int KT = K / KSTEP;
  issue(0, 0);
  for (int kt = 0; kt < KT; ++kt) {
    const int cur = kt & 1;
    wait_async0();        // this wave's copies for tile kt are in LDS
    __syncthreads();      // ... and so are every other wave's
    if (kt + 1 < KT) issue(cur ^ 1, (kt + 1) * KSTEP);  // overlap with compute

    const _Float16* ap = &As[cur][m0l + (lane & 15)][(lane >> 4) << 3];
    v16h a = cat16(*(const h8*)ap, *(const h8*)(ap + 16));
#pragma unroll
    for (int j = 0; j < 4; ++j) {
      const _Float16* bp = &Bs[cur][n0l + 16 * j + (lane & 15)][(lane >> 4) << 3];
      v16h b = cat16(*(const h8*)bp, *(const h8*)(bp + 16));
      acc[j] = wmma16(a, b, acc[j]);
    }
  }

  const int mrow  = mt + m0l + ((lane >> 4) << 3);
  const int ncol0 = nt + n0l + (lane & 15);
#pragma unroll
  for (int j = 0; j < 4; ++j) {
    const int nn = ncol0 + 16 * j;
    const float bv = bias[nn];
#pragma unroll
    for (int r = 0; r < 8; ++r) {
      const int mm = mrow + r;
      const float val = acc[j][r] + bv;
      if (mode == 3) {
        out_f[(size_t)mm * (size_t)N + nn] = val;
      } else {
        const int b  = mm / Tt, tt = mm % Tt;
        const int hh = nn >> 7, d = nn & (HD - 1);
        const size_t bh = (size_t)b * Hh + hh;
        if (mode == 2)
          out_h[(bh * HD + d) * Tt + tt] = (_Float16)val;     // V^T [B,H,hd,T]
        else
          out_h[(bh * Tt + tt) * HD + d] = (_Float16)val;     // [B,H,T,hd]
      }
    }
  }
}

// ---------------------------------------------------------------------------
// Kernel 2: logits = Q K^T / sqrt(hd) with causal mask -> probs buffer (f32)
// grid: (Tk/128, Tq/64, B*H). K=128 -> only 4 steps; direct-from-L2 loads.
// ---------------------------------------------------------------------------
__global__ void __launch_bounds__(256)
attn_logits(const _Float16* __restrict__ q16, const _Float16* __restrict__ k16,
            float* __restrict__ probs) {
  const int lane = threadIdx.x & 31;
  const int w    = threadIdx.x >> 5;
  const int bh   = blockIdx.z;
  const int q0t  = blockIdx.y * 64;
  const int k0t  = blockIdx.x * 128;
  float* pt = probs + (size_t)bh * Tt * Tt;

  if (k0t > q0t + 63) {                 // tile entirely above the diagonal
    for (int i = threadIdx.x; i < 64 * 128; i += 256) {
      const int rq = i >> 7, rk = i & 127;
      pt[(size_t)(q0t + rq) * Tt + (k0t + rk)] = -__builtin_inff();
    }
    return;
  }

  const _Float16* qb = q16 + (size_t)bh * Tt * HD;
  const _Float16* kb = k16 + (size_t)bh * Tt * HD;
  const int m0 = q0t + (w & 3) * 16;
  const int n0 = k0t + (w >> 2) * 64;

  v8f acc[4] = {};
#pragma unroll
  for (int k0 = 0; k0 < HD; k0 += 32) {
    v16h a = load_a_f16(qb, HD, m0, k0, lane);
#pragma unroll
    for (int j = 0; j < 4; ++j) {
      v16h b = load_b_f16(kb, HD, n0 + 16 * j, k0, lane);
      acc[j] = wmma16(a, b, acc[j]);
    }
  }

  const float scale = 0.08838834764831845f;   // 1/sqrt(128)
  const int mrow  = m0 + ((lane >> 4) << 3);
  const int ncol0 = n0 + (lane & 15);
#pragma unroll
  for (int j = 0; j < 4; ++j) {
    const int nn = ncol0 + 16 * j;
#pragma unroll
    for (int r = 0; r < 8; ++r) {
      const int mm = mrow + r;
      pt[(size_t)mm * Tt + nn] = (nn <= mm) ? acc[j][r] * scale : -__builtin_inff();
    }
  }
}

// ---------------------------------------------------------------------------
// Kernel 3: row softmax in place; one 256-thread block per row of 2048
// ---------------------------------------------------------------------------
__global__ void __launch_bounds__(256)
softmax_rows(float* __restrict__ probs) {
  __shared__ float red[256];
  const int tid = threadIdx.x;
  float* p = probs + (size_t)blockIdx.x * Tt;

  float v[8];
  float m = -__builtin_inff();
#pragma unroll
  for (int i = 0; i < 8; ++i) { v[i] = p[tid + (i << 8)]; m = fmaxf(m, v[i]); }
  red[tid] = m; __syncthreads();
  for (int s = 128; s > 0; s >>= 1) {
    if (tid < s) red[tid] = fmaxf(red[tid], red[tid + s]);
    __syncthreads();
  }
  const float rowmax = red[0]; __syncthreads();

  float e[8]; float sum = 0.f;
#pragma unroll
  for (int i = 0; i < 8; ++i) { e[i] = __expf(v[i] - rowmax); sum += e[i]; }
  red[tid] = sum; __syncthreads();
  for (int s = 128; s > 0; s >>= 1) {
    if (tid < s) red[tid] += red[tid + s];
    __syncthreads();
  }
  const float inv = 1.0f / red[0];
#pragma unroll
  for (int i = 0; i < 8; ++i) p[tid + (i << 8)] = e[i] * inv;
}

// ---------------------------------------------------------------------------
// Kernel 5: ctx = P * V  (A = probs f32 -> f16 on load, B = V^T rows)
// grid: (1, Tq/64, B*H); skips k-tiles past the diagonal (probs there == 0)
// ---------------------------------------------------------------------------
__global__ void __launch_bounds__(256)
attn_context(const float* __restrict__ probs, const _Float16* __restrict__ vT,
             _Float16* __restrict__ ctx16) {
  const int lane = threadIdx.x & 31;
  const int w    = threadIdx.x >> 5;
  const int bh   = blockIdx.z;
  const int q0t  = blockIdx.y * 64;

  const float*    pb = probs + (size_t)bh * Tt * Tt;
  const _Float16* vb = vT    + (size_t)bh * HD * Tt;

  const int m0 = q0t + (w & 3) * 16;
  const int n0 = (w >> 2) * 64;                 // hd split across 2 waves

  v8f acc[4] = {};
  const int kend = q0t + 64;                    // causal: probs beyond are 0
  for (int k0 = 0; k0 < kend; k0 += 32) {
    v16h a = load_a_f32(pb, Tt, m0, k0, lane);
#pragma unroll
    for (int j = 0; j < 4; ++j) {
      v16h b = load_b_f16(vb, Tt, n0 + 16 * j, k0, lane);
      acc[j] = wmma16(a, b, acc[j]);
    }
  }

  const int b  = bh >> 3;
  const int hh = bh & 7;
  const int mrow  = m0 + ((lane >> 4) << 3);
  const int ncol0 = n0 + (lane & 15);
#pragma unroll
  for (int j = 0; j < 4; ++j) {
    const int nn = ncol0 + 16 * j;
#pragma unroll
    for (int r = 0; r < 8; ++r) {
      const int mm = mrow + r;
      ctx16[((size_t)b * Tt + mm) * Dm + (hh * HD + nn)] = (_Float16)acc[j][r];
    }
  }
}

// ---------------------------------------------------------------------------
extern "C" void kernel_launch(void* const* d_in, const int* in_sizes, int n_in,
                              void* d_out, int out_size, void* d_ws, size_t ws_size,
                              hipStream_t stream) {
  (void)in_sizes; (void)n_in; (void)out_size; (void)ws_size;
  const float* x  = (const float*)d_in[0];
  const float* y  = (const float*)d_in[1];
  const float* wq = (const float*)d_in[2];
  const float* bq = (const float*)d_in[3];
  const float* wk = (const float*)d_in[4];
  const float* bk = (const float*)d_in[5];
  const float* wv = (const float*)d_in[6];
  const float* bv = (const float*)d_in[7];
  const float* wo = (const float*)d_in[8];
  const float* bo = (const float*)d_in[9];

  float* out = (float*)d_out;
  const size_t Sx = (size_t)Bb * Tt * Dm;         // 8,388,608
  float* probs = out + Sx;                        // [B,H,Tq,Tk] f32 output #2

  _Float16* ws    = (_Float16*)d_ws;
  _Float16* x16   = ws;
  _Float16* y16   = x16  + Sx;
  _Float16* wq16  = y16  + Sx;
  _Float16* wk16  = wq16 + (size_t)Dm * Dm;
  _Float16* wv16  = wk16 + (size_t)Dm * Dm;
  _Float16* wo16  = wv16 + (size_t)Dm * Dm;
  _Float16* q16   = wo16 + (size_t)Dm * Dm;       // [B,H,Tq,hd]
  _Float16* k16   = q16  + Sx;                    // [B,H,Tk,hd]
  _Float16* vT16  = k16  + Sx;                    // [B,H,hd,Tk]
  _Float16* ctx16 = vT16 + Sx;                    // [B,Tq,D]

  // 0) precision convert
  f32_to_f16<<<2048, 256, 0, stream>>>(x,  x16,  Sx);
  f32_to_f16<<<2048, 256, 0, stream>>>(y,  y16,  Sx);
  f32_to_f16<<<512,  256, 0, stream>>>(wq, wq16, (size_t)Dm * Dm);
  f32_to_f16<<<512,  256, 0, stream>>>(wk, wk16, (size_t)Dm * Dm);
  f32_to_f16<<<512,  256, 0, stream>>>(wv, wv16, (size_t)Dm * Dm);
  f32_to_f16<<<512,  256, 0, stream>>>(wo, wo16, (size_t)Dm * Dm);

  // 1) QKV projections (M = B*T = 8192, N = K = 1024)
  dim3 gp((Bb * Tt) / 64, Dm / 128);
  gemm_proj<<<gp, 256, 0, stream>>>(x16, wq16, bq, q16,  nullptr, Dm, Dm, 0);
  gemm_proj<<<gp, 256, 0, stream>>>(y16, wk16, bk, k16,  nullptr, Dm, Dm, 0);
  gemm_proj<<<gp, 256, 0, stream>>>(y16, wv16, bv, vT16, nullptr, Dm, Dm, 2);

  // 2) causal logits
  dim3 gl(Tt / 128, Tt / 64, Bb * Hh);
  attn_logits<<<gl, 256, 0, stream>>>(q16, k16, probs);

  // 3) softmax (normalizes probs output in place)
  softmax_rows<<<Bb * Hh * Tt, 256, 0, stream>>>(probs);

  // 4) context = P*V
  dim3 gc(1, Tt / 64, Bb * Hh);
  attn_context<<<gc, 256, 0, stream>>>(probs, vT16, ctx16);

  // 5) output projection (f32 + bias)
  gemm_proj<<<gp, 256, 0, stream>>>(ctx16, wo16, bo, nullptr, out, Dm, Dm, 3);
}
